// Model_86835648791136
// MI455X (gfx1250) — matrix operand order repaired and checked by reading, not compiled, not run on hardware
//
#include <hip/hip_runtime.h>
#include <hip/hip_bf16.h>
#include <stdint.h>

// ---------------------------------------------------------------------------
// VQ-VAE recommender forward pass for gfx1250 (MI455X).
// Conv/GEMM layers: v_wmma_f32_16x16x32_bf16, activations bf16 [B][L][C],
// weights pre-packed into B-fragment order, double-buffered LDS staging
// via GLOBAL_LOAD_ASYNC_TO_LDS_B128 (ASYNCcnt) when available.
// VQ distance search: WMMA dots + cross-lane shuffle argmin.
// ---------------------------------------------------------------------------

typedef __bf16 bf16;
typedef __attribute__((ext_vector_type(16))) __bf16 v16bf;
typedef __attribute__((ext_vector_type(8)))  float  v8f;
typedef int aint4 __attribute__((vector_size(16)));   // builtin's pointer elem type

#if defined(__AMDGCN__) && __has_builtin(__builtin_amdgcn_global_load_async_to_lds_b128) && __has_builtin(__builtin_amdgcn_s_wait_asynccnt)
#define USE_ASYNC_LDS 1
#else
#define USE_ASYNC_LDS 0
#endif

#define DEVINL __device__ __forceinline__

static constexpr int B_     = 128;
static constexpr int NITEMS = 8192;
static constexpr int LENC   = 4096;
static constexpr int HID    = 256;
static constexpr int LAT    = 64;
static constexpr int NEMB   = 512;
static constexpr int NQ     = 4;
static constexpr int MROWS  = (B_ * LAT * LENC) / LAT;   // 524288 VQ rows of 64
static constexpr int KSPLIT = 8;                         // out-proj split-K

union FragU {
    uint4 q[2];
    v16bf v;
};

// A-matrix (16x32, 16-bit) half->K mapping from CDNA5 ISA 7.12.2
DEVINL int kmapA(int h, int hi) { return h + (h & 8) + (hi << 3); }
// B-matrix (32x16, 16-bit) half->K mapping (same table order as A)
DEVINL int kmapB(int h, int hi) { return h + (h & 8) + (hi << 3); }

// ---------------------------------------------------------------------------
// Weight packing: src(co, ci, tap) -> B fragments, frag index
//   f = ((chunk*NT + tap)*nCoTiles + coTile), 512 bf16 per fragment,
//   lane-major inside (lane holds 16 halves = 32 bytes contiguous).
// ---------------------------------------------------------------------------
__global__ void pack_weights_kernel(const float* __restrict__ src,
                                    bf16* __restrict__ dst,
                                    int nCoTiles, int nChunks, int NT,
                                    int coStride, int ciStride, int4 tapOff)
{
    int t = blockIdx.x * blockDim.x + threadIdx.x;
    int total = nCoTiles * nChunks * NT * 32;
    if (t >= total) return;
    int lane = t & 31;
    int f    = t >> 5;
    int ct   = f % nCoTiles;
    int r    = f / nCoTiles;
    int tap  = r % NT;
    int ch   = r / NT;
    int toff = (tap == 0) ? tapOff.x : (tap == 1) ? tapOff.y
             : (tap == 2) ? tapOff.z : tapOff.w;
    int lo = lane & 15, hi = lane >> 4;
    int co = ct * 16 + lo;
    FragU u;
#pragma unroll
    for (int h = 0; h < 16; ++h) {
        int ci = ch * 32 + kmapB(h, hi);
        u.v[h] = (bf16)src[(size_t)co * coStride + (size_t)ci * ciStride + toff];
    }
    uint4* d = (uint4*)dst + ((size_t)f << 6) + (lane << 1);
    d[0] = u.q[0];
    d[1] = u.q[1];
}

// ---------------------------------------------------------------------------
// Generic WMMA conv / GEMM kernel, software-pipelined weight staging.
//   out[b, n*OSTR+OOFF, co] = act( sum_taps sum_ci W[co,ci,tap] *
//                                  X[b, n+ioff[tap], ci] + bias[co] (+res) )
// WG = 256 threads = 8 waves; wave w owns positions n0 = (bx*8+w)*16.
// All waves share co range blockIdx.y*16G .. +16G (G co-tiles per wave).
// OUTMODE: 0 = bf16 [B][LOUT][COUT] (LDS-bounced vector stores)
//          1 = f32 [B][COUT][LOUT] scalar stores
//          2 = f32 split-K partial [split][LOUT][COUT] stores (b==0)
// nSplit>1: blockIdx.z = K split index (b forced 0), each split does
//           nChunks/nSplit chunks.
// ---------------------------------------------------------------------------
template <int G, int OUTMODE>
__global__ void conv_wmma_kernel(const bf16* __restrict__ X,
                                 const bf16* __restrict__ Wpack,
                                 const float* __restrict__ bias,
                                 const bf16* __restrict__ Res,
                                 void* __restrict__ Out,
                                 int CIN, int LIN, int LOUT, int COUTtot,
                                 int NT, int4 ioff, int OSTR, int OOFF,
                                 int relu, int nSplit)
{
    extern __shared__ char smemRaw[];
    uint4* sbuf0 = (uint4*)smemRaw;          // G * 64 uint4 each
    uint4* sbuf1 = sbuf0 + G * 64;

    const int tid  = threadIdx.x;
    const int wave = tid >> 5;
    const int lane = tid & 31;
    const int lo   = lane & 15;
    const int hi   = lane >> 4;

    const int split    = (nSplit > 1) ? blockIdx.z : 0;
    const int b        = (nSplit > 1) ? 0 : blockIdx.z;
    const int n0       = (blockIdx.x * 8 + wave) * 16;
    const int ct0      = blockIdx.y * G;
    const int nCoTiles = COUTtot >> 4;
    const int nChunks  = CIN >> 5;
    const int chkPer   = (nSplit > 1) ? (nChunks / nSplit) : nChunks;
    const int s0       = NT * chkPer * split;
    const int S        = s0 + NT * chkPer;

    const v8f vzero = {0.f, 0.f, 0.f, 0.f, 0.f, 0.f, 0.f, 0.f};
    v8f acc[G];
#pragma unroll
    for (int g = 0; g < G; ++g) acc[g] = vzero;

    const int aRow = n0 + lo;   // output position handled by this lane's A row
    __builtin_prefetch(X + ((size_t)b * LIN + aRow) * CIN, 0, 1);

#if USE_ASYNC_LDS
    auto stageAsync = [&](int s, uint4* dst) {
        const uint4* src = (const uint4*)Wpack + (((size_t)s * nCoTiles + ct0) << 6);
#pragma unroll
        for (int i = 0; i < (G * 64 + 255) / 256; ++i) {
            int idx = tid + i * 256;
            if (idx < G * 64)
                __builtin_amdgcn_global_load_async_to_lds_b128(
                    (aint4*)(src + idx), (aint4*)(dst + idx), 0, 0);
        }
    };
    stageAsync(s0, sbuf0);
    __builtin_amdgcn_s_wait_asynccnt(0);
    __syncthreads();
#else
    constexpr int NR = (G * 64 + 255) / 256;   // staging uint4 per thread
    uint4 streg[NR];
    auto loadSlice = [&](int s) {
        const uint4* src = (const uint4*)Wpack + (((size_t)s * nCoTiles + ct0) << 6);
#pragma unroll
        for (int i = 0; i < NR; ++i) {
            int idx = tid + i * 256;
            if (idx < G * 64) streg[i] = src[idx];
        }
    };
    auto storeSlice = [&](uint4* dst) {
#pragma unroll
        for (int i = 0; i < NR; ++i) {
            int idx = tid + i * 256;
            if (idx < G * 64) dst[idx] = streg[i];
        }
    };
    loadSlice(s0);
    storeSlice(sbuf0);
    __syncthreads();
#endif

    for (int s = s0; s < S; ++s) {
        uint4* cur = (s & 1) ? sbuf1 : sbuf0;
        uint4* nxt = (s & 1) ? sbuf0 : sbuf1;
#if USE_ASYNC_LDS
        if (s + 1 < S) stageAsync(s + 1, nxt);   // DMA in background
#else
        if (s + 1 < S) loadSlice(s + 1);         // in flight under the math
#endif
        const int ch  = s / NT;
        const int tap = s - ch * NT;
        const int io  = (tap == 0) ? ioff.x : (tap == 1) ? ioff.y
                      : (tap == 2) ? ioff.z : ioff.w;

        // A fragment: 16 positions x 32 channels from global (bf16 NLC)
        FragU fa;
        int ip = aRow + io;
        if (ip >= 0 && ip < LIN) {
            const uint4* ap = (const uint4*)(X +
                (((size_t)b * LIN + ip) * CIN + (ch << 5) + (hi << 3)));
            fa.q[0] = ap[0];
            fa.q[1] = ap[2];     // +16 halves
        } else {
            fa.q[0] = make_uint4(0u, 0u, 0u, 0u);
            fa.q[1] = make_uint4(0u, 0u, 0u, 0u);
        }

#pragma unroll
        for (int g = 0; g < G; ++g) {
            FragU fb;
            const uint4* bp = cur + ((g << 6) + (lane << 1));
            fb.q[0] = bp[0];
            fb.q[1] = bp[1];
            acc[g] = __builtin_amdgcn_wmma_f32_16x16x32_bf16(
                false, fa.v, false, fb.v, (short)0, acc[g], false, false);
        }

#if USE_ASYNC_LDS
        __builtin_amdgcn_s_wait_asynccnt(0);
#else
        if (s + 1 < S) storeSlice(nxt);
#endif
        __syncthreads();
    }

    if constexpr (OUTMODE == 0) {
        // bounce acc+bias through LDS, then vectorized res-add/relu/store
        bf16* sOut = (bf16*)smemRaw;
        const int WGC = 16 * G;                       // WG co width
#pragma unroll
        for (int g = 0; g < G; ++g) {
            const int co = (ct0 + g) * 16 + lo;
            float bs = bias ? bias[co] : 0.f;
#pragma unroll
            for (int r = 0; r < 8; ++r)
                sOut[wave * 16 * WGC + (r + (hi << 3)) * WGC + g * 16 + lo] =
                    (bf16)(acc[g][r] + bs);
        }
        __syncthreads();
        const int perWaveU4 = 32 * G, rowU4 = 2 * G;
        for (int j = tid; j < 8 * perWaveU4; j += 256) {
            int w  = j / perWaveU4;
            int rm = j - w * perWaveU4;
            int r  = rm / rowU4;
            int c8 = (rm - r * rowU4) << 3;           // half offset in WG co range
            int n  = (blockIdx.x * 8 + w) * 16 + r;
            int op = n * OSTR + OOFF;
            size_t dstIdx = ((size_t)b * LOUT + op) * COUTtot + ct0 * 16 + c8;
            union { uint4 q; bf16 h[8]; } o, rr;
            o.q = *(const uint4*)(sOut + ((size_t)j << 3));
            if (Res) {
                rr.q = *(const uint4*)(Res + dstIdx);
#pragma unroll
                for (int k = 0; k < 8; ++k)
                    o.h[k] = (bf16)((float)o.h[k] + (float)rr.h[k]);
            }
            if (relu) {
#pragma unroll
                for (int k = 0; k < 8; ++k)
                    o.h[k] = (bf16)fmaxf((float)o.h[k], 0.f);
            }
            *(uint4*)((bf16*)Out + dstIdx) = o.q;
        }
    } else {
#pragma unroll
        for (int g = 0; g < G; ++g) {
            const int co = (ct0 + g) * 16 + lo;
            float bs = bias ? bias[co] : 0.f;
#pragma unroll
            for (int r = 0; r < 8; ++r) {
                int n  = n0 + r + (hi << 3);
                int op = n * OSTR + OOFF;
                float v = acc[g][r] + bs;
                if (relu) v = fmaxf(v, 0.f);
                if constexpr (OUTMODE == 1) {
                    ((float*)Out)[((size_t)b * COUTtot + co) * LOUT + op] = v;
                } else {   // OUTMODE 2: split-K partial (b==0)
                    ((float*)Out)[((size_t)split * LOUT + op) * COUTtot + co] = v;
                }
            }
        }
    }
}

// reduce split-K partials + bias -> final recon
__global__ void reduce_out_kernel(const float* __restrict__ part,
                                  const float* __restrict__ bias,
                                  float* __restrict__ out)
{
    size_t t = (size_t)blockIdx.x * blockDim.x + threadIdx.x;   // B_*NITEMS
    int co = (int)(t & (NITEMS - 1));
    float a = bias[co];
#pragma unroll
    for (int s = 0; s < KSPLIT; ++s) a += part[(size_t)s * B_ * NITEMS + t];
    out[t] = a;
}

// ---------------------------------------------------------------------------
// Encoder stem: gather user row + Conv1d(1->256, k=4, s=2, p=1) + ReLU.
// ---------------------------------------------------------------------------
__global__ void enc_conv_kernel(const int* __restrict__ uids,
                                const float* __restrict__ imat,
                                const float* __restrict__ w,   // [256][1][4]
                                const float* __restrict__ bias,
                                bf16* __restrict__ out)        // [B][4096][256]
{
    __shared__ float sw[HID * 4];
    __shared__ float sb[HID];
    for (int i = threadIdx.x; i < HID * 4; i += blockDim.x) sw[i] = w[i];
    for (int i = threadIdx.x; i < HID; i += blockDim.x) sb[i] = bias[i];
    __syncthreads();

    int t   = blockIdx.x * blockDim.x + threadIdx.x;
    int cog = t & 31;
    int l   = (t >> 5) & (LENC - 1);
    int b   = t >> 17;
    const float* xrow = imat + (size_t)uids[b] * NITEMS;
    float x[4];
#pragma unroll
    for (int k = 0; k < 4; ++k) {
        int p = 2 * l + k - 1;
        x[k] = (p >= 0 && p < NITEMS) ? xrow[p] : 0.f;
    }
    bf16 o[8];
#pragma unroll
    for (int j = 0; j < 8; ++j) {
        int co = cog * 8 + j;
        float a = sb[co];
#pragma unroll
        for (int k = 0; k < 4; ++k) a += sw[co * 4 + k] * x[k];
        o[j] = (bf16)fmaxf(a, 0.f);
    }
    *(uint4*)(out + ((size_t)b * LENC + l) * HID + cog * 8) = *(uint4*)o;
}

// codebook row norms, all stages at once
__global__ void cb_norms_kernel(const float* __restrict__ cbs, float* __restrict__ norms)
{
    int c = blockIdx.x * blockDim.x + threadIdx.x;     // over NQ*NEMB
    const float* row = cbs + (size_t)c * LAT;
    float s = 0.f;
#pragma unroll
    for (int i = 0; i < LAT; i += 4) {
        float4 v = *(const float4*)(row + i);
        s += v.x * v.x + v.y * v.y + v.z * v.z + v.w * v.w;
    }
    norms[c] = s;
}

// ---------------------------------------------------------------------------
// One residual-VQ stage with WMMA distance search.
// ---------------------------------------------------------------------------
__global__ void vq_wmma_kernel(const float* __restrict__ codebook, // [512][64] f32
                               const bf16*  __restrict__ cbPack,   // packed B frags
                               const float* __restrict__ normsG,   // [512]
                               float* __restrict__ residual,
                               float* __restrict__ zq,
                               float* __restrict__ counts,
                               float* __restrict__ lossAcc)
{
    extern __shared__ char smemRaw[];
    uint4* sCB   = (uint4*)smemRaw;                       // 4096 uint4 = 64KB
    float* snorm = (float*)(smemRaw + 65536);             // 512 f32
    float* sbV   = snorm + NEMB;                          // 128 f32 (8 waves * 16)
    int*   sbI   = (int*)(sbV + 128);                     // 128 int
    float* sred  = (float*)(sbI + 128);                   // 256 f32

    const int tid = threadIdx.x, wave = tid >> 5, lane = tid & 31;
    const int lo = lane & 15, hi = lane >> 4;

#if USE_ASYNC_LDS
    for (int i = tid; i < 4096; i += 256)
        __builtin_amdgcn_global_load_async_to_lds_b128(
            (aint4*)((const uint4*)cbPack + i), (aint4*)(sCB + i), 0, 0);
    for (int c = tid; c < NEMB; c += 256) snorm[c] = normsG[c];
    __builtin_amdgcn_s_wait_asynccnt(0);
#else
    for (int i = tid; i < 4096; i += 256) sCB[i] = ((const uint4*)cbPack)[i];
    for (int c = tid; c < NEMB; c += 256) snorm[c] = normsG[c];
#endif
    __syncthreads();

    const size_t row0 = ((size_t)blockIdx.x * 8 + wave) * 16;

    // A fragments (16 rows x 32 dims, two K chunks) from fp32 residual
    FragU fa[2];
#pragma unroll
    for (int ch = 0; ch < 2; ++ch) {
        const float* rp = residual + (row0 + lo) * LAT + ch * 32 + (hi << 3);
        float4 a0 = *(const float4*)(rp);
        float4 a1 = *(const float4*)(rp + 4);
        float4 b0 = *(const float4*)(rp + 16);
        float4 b1 = *(const float4*)(rp + 20);
        fa[ch].v[0]  = (bf16)a0.x; fa[ch].v[1]  = (bf16)a0.y;
        fa[ch].v[2]  = (bf16)a0.z; fa[ch].v[3]  = (bf16)a0.w;
        fa[ch].v[4]  = (bf16)a1.x; fa[ch].v[5]  = (bf16)a1.y;
        fa[ch].v[6]  = (bf16)a1.z; fa[ch].v[7]  = (bf16)a1.w;
        fa[ch].v[8]  = (bf16)b0.x; fa[ch].v[9]  = (bf16)b0.y;
        fa[ch].v[10] = (bf16)b0.z; fa[ch].v[11] = (bf16)b0.w;
        fa[ch].v[12] = (bf16)b1.x; fa[ch].v[13] = (bf16)b1.y;
        fa[ch].v[14] = (bf16)b1.z; fa[ch].v[15] = (bf16)b1.w;
    }

    const v8f vzero = {0.f, 0.f, 0.f, 0.f, 0.f, 0.f, 0.f, 0.f};
    float bestv[8];
    int   besti[8];
#pragma unroll
    for (int r = 0; r < 8; ++r) { bestv[r] = 3.4e38f; besti[r] = 0; }

    for (int t4 = 0; t4 < 8; ++t4) {          // 8 groups of 4 code tiles
        v8f acc[4];
#pragma unroll
        for (int t = 0; t < 4; ++t) {
            acc[t] = vzero;
            const int tile = t4 * 4 + t;
#pragma unroll
            for (int ch = 0; ch < 2; ++ch) {
                FragU fb;
                const uint4* bp = sCB + (((ch * 32 + tile) << 6) + (lane << 1));
                fb.q[0] = bp[0];
                fb.q[1] = bp[1];
                acc[t] = __builtin_amdgcn_wmma_f32_16x16x32_bf16(
                    false, fa[ch].v, false, fb.v, (short)0, acc[t], false, false);
            }
        }
#pragma unroll
        for (int t = 0; t < 4; ++t) {
            const int c = (t4 * 4 + t) * 16 + lo;
            const float nrm = snorm[c];
#pragma unroll
            for (int r = 0; r < 8; ++r) {
                float d = nrm - 2.f * acc[t][r];
                if (d < bestv[r]) { bestv[r] = d; besti[r] = c; }
            }
        }
    }

    // cross-lane argmin inside each 16-lane half (first index wins ties)
#pragma unroll
    for (int r = 0; r < 8; ++r) {
        float v = bestv[r];
        int   ii = besti[r];
#pragma unroll
        for (int m = 1; m < 16; m <<= 1) {
            float ov = __shfl_xor(v, m, 32);
            int   oi = __shfl_xor(ii, m, 32);
            if (ov < v || (ov == v && oi < ii)) { v = ov; ii = oi; }
        }
        bestv[r] = v; besti[r] = ii;
    }
    if (lo == 0) {
#pragma unroll
        for (int r = 0; r < 8; ++r) {
            int m = (hi << 3) + r;
            sbV[wave * 16 + m] = bestv[r];
            sbI[wave * 16 + m] = besti[r];
        }
    }
    __syncthreads();

    // update: lane j handles row j/2, dim half j&1 (32 dims), exact fp32 SSE
    float sse = 0.f;
    {
        const int m = lane >> 1, half = lane & 1;
        const int bi = sbI[wave * 16 + m];
        const float* cb = codebook + (size_t)bi * LAT + half * 32;
        float* rp = residual + (row0 + m) * LAT + half * 32;
        float* zp = zq + (row0 + m) * LAT + half * 32;
#pragma unroll
        for (int i = 0; i < 32; i += 4) {
            float4 f = *(const float4*)(rp + i);
            float4 q = *(const float4*)(cb + i);
            float4 z = *(const float4*)(zp + i);
            float4 d = make_float4(f.x - q.x, f.y - q.y, f.z - q.z, f.w - q.w);
            sse += d.x * d.x + d.y * d.y + d.z * d.z + d.w * d.w;
            z.x += q.x; z.y += q.y; z.z += q.z; z.w += q.w;
            *(float4*)(zp + i) = z;
            *(float4*)(rp + i) = d;
        }
        if (half == 0) atomicAdd(counts + bi, 1.f);
    }
    sred[tid] = sse;
    __syncthreads();
    for (int s = 128; s > 0; s >>= 1) {
        if (tid < s) sred[tid] += sred[tid + s];
        __syncthreads();
    }
    if (tid == 0) atomicAdd(lossAcc, sred[0]);
}

// zq f32 [B][64][4096] -> bf16 [B][4096][64]
__global__ void zq_to_nlc_kernel(const float* __restrict__ zq, bf16* __restrict__ out)
{
    size_t t = (size_t)blockIdx.x * blockDim.x + threadIdx.x;
    int l   = (int)(t & (LENC - 1));
    int lat = (int)((t >> 12) & 63);
    int b   = (int)(t >> 18);
    out[((size_t)b * LENC + l) * LAT + lat] = (bf16)zq[t];
}

// dec_finalT: ConvTranspose1d(256->1, k=3, s=1, p=1); out as bf16 row for GEMM A
__global__ void dec_finalT_kernel(const bf16* __restrict__ X,   // [B][8192][256]
                                  const float* __restrict__ w,  // [256][1][3]
                                  const float* __restrict__ bias,
                                  bf16* __restrict__ dbf)       // [B][8192]
{
    __shared__ float sw[HID * 3];
    for (int i = threadIdx.x; i < HID * 3; i += blockDim.x) sw[i] = w[i];
    __syncthreads();

    size_t t = (size_t)blockIdx.x * blockDim.x + threadIdx.x;
    int j = (int)(t & (NITEMS - 1));
    int b = (int)(t >> 13);
    float a = bias[0];
#pragma unroll
    for (int tt = 0; tt < 3; ++tt) {
        int l = j + 1 - tt;
        if (l < 0 || l >= NITEMS) continue;
        const bf16* xr = X + ((size_t)b * NITEMS + l) * HID;
        for (int ci = 0; ci < HID; ci += 8) {
            union { uint4 q; bf16 h[8]; } u;
            u.q = *(const uint4*)(xr + ci);
#pragma unroll
            for (int k = 0; k < 8; ++k) a += sw[(ci + k) * 3 + tt] * (float)u.h[k];
        }
    }
    dbf[t] = (bf16)a;
}

__global__ void finalize_kernel(const float* __restrict__ counts,
                                const float* __restrict__ lossAcc,
                                float* __restrict__ out)
{
    __shared__ float sred[256];
    float psum = 0.f;
    for (int q = 0; q < NQ; ++q) {
        float tsum = 0.f;
        for (int c = threadIdx.x; c < NEMB; c += 256) {
            float avg = counts[q * NEMB + c] * (1.f / 524288.f);
            tsum += avg * logf(avg + 1e-10f);
        }
        sred[threadIdx.x] = tsum;
        __syncthreads();
        for (int s = 128; s > 0; s >>= 1) {
            if ((int)threadIdx.x < s) sred[threadIdx.x] += sred[threadIdx.x + s];
            __syncthreads();
        }
        if (threadIdx.x == 0) psum += expf(-sred[0]);
        __syncthreads();
    }
    if (threadIdx.x == 0) {
        out[0] = 1.25f * lossAcc[0] / 33554432.f;   // (1+CC)*SSE/numel, summed stages
        out[1] = psum * (1.f / NQ);
    }
}

// ---------------------------------------------------------------------------
extern "C" void kernel_launch(void* const* d_in, const int* in_sizes, int n_in,
                              void* d_out, int out_size, void* d_ws, size_t ws_size,
                              hipStream_t stream)
{
    (void)in_sizes; (void)n_in; (void)out_size; (void)ws_size;

    const int*   user_ids    = (const int*)  d_in[0];
    const float* imat        = (const float*)d_in[1];
    const float* enc_conv_w  = (const float*)d_in[2];
    const float* enc_conv_b  = (const float*)d_in[3];
    const float* enc_res_w1  = (const float*)d_in[4];
    const float* enc_res_b1  = (const float*)d_in[5];
    const float* enc_res_w2  = (const float*)d_in[6];
    const float* enc_res_b2  = (const float*)d_in[7];
    const float* enc_final_w = (const float*)d_in[8];
    const float* enc_final_b = (const float*)d_in[9];
    const float* codebooks   = (const float*)d_in[10];
    const float* dec_convT_w = (const float*)d_in[11];
    const float* dec_convT_b = (const float*)d_in[12];
    const float* dec_res_w1  = (const float*)d_in[13];
    const float* dec_res_b1  = (const float*)d_in[14];
    const float* dec_res_w2  = (const float*)d_in[15];
    const float* dec_res_b2  = (const float*)d_in[16];
    const float* dec_fin_w   = (const float*)d_in[17];
    const float* dec_fin_b   = (const float*)d_in[18];
    const float* out_proj_w  = (const float*)d_in[19];
    const float* out_proj_b  = (const float*)d_in[20];
    float* out = (float*)d_out;

    char* ws = (char*)d_ws;
    size_t off = 0;
    auto alloc = [&](size_t bytes) -> char* {
        char* p = ws + off;
        off += (bytes + 255) & ~(size_t)255;
        return p;
    };
    bf16*  bufA     = (bf16*) alloc((size_t)B_ * NITEMS * HID * 2);
    bf16*  bufB     = (bf16*) alloc((size_t)B_ * NITEMS * HID * 2);
    float* residual = (float*)alloc((size_t)B_ * LAT * LENC * 4);
    float* zq       = (float*)alloc((size_t)B_ * LAT * LENC * 4);
    bf16*  zqN      = (bf16*) alloc((size_t)B_ * LENC * LAT * 2);
    bf16*  dbf      = (bf16*) alloc((size_t)B_ * NITEMS * 2);
    float* partOut  = (float*)alloc((size_t)KSPLIT * B_ * NITEMS * 4);
    bf16*  pE1[2]   = { (bf16*)alloc(393216), (bf16*)alloc(393216) };
    bf16*  pE2[2]   = { (bf16*)alloc(131072), (bf16*)alloc(131072) };
    bf16*  pEF      = (bf16*) alloc(98304);
    bf16*  pDTe     = (bf16*) alloc(65536);
    bf16*  pDTo     = (bf16*) alloc(65536);
    bf16*  pD1[2]   = { (bf16*)alloc(393216), (bf16*)alloc(393216) };
    bf16*  pD2[2]   = { (bf16*)alloc(131072), (bf16*)alloc(131072) };
    bf16*  pW       = (bf16*) alloc((size_t)134217728);
    bf16*  pCB      = (bf16*) alloc((size_t)NQ * 65536);       // packed codebooks
    float* cbNorms  = (float*)alloc(NQ * NEMB * 4);
    float* counts   = (float*)alloc((NQ * NEMB + 4) * 4);
    float* lossAcc  = counts + NQ * NEMB;

    (void)hipMemsetAsync(zq, 0, (size_t)B_ * LAT * LENC * 4, stream);
    (void)hipMemsetAsync(counts, 0, (NQ * NEMB + 4) * 4, stream);

    auto pack = [&](const float* src, bf16* dst, int nct, int nch, int nt,
                    int cs, int cis, int4 toff) {
        int total = nct * nch * nt * 32;
        pack_weights_kernel<<<(total + 255) / 256, 256, 0, stream>>>(
            src, dst, nct, nch, nt, cs, cis, toff);
    };
    for (int i = 0; i < 2; ++i) {
        pack(enc_res_w1 + (size_t)i * HID * HID * 3, pE1[i], 16, 8, 3, HID * 3, 3, make_int4(0, 1, 2, 0));
        pack(enc_res_w2 + (size_t)i * HID * HID,     pE2[i], 16, 8, 1, HID,     1, make_int4(0, 0, 0, 0));
        pack(dec_res_w1 + (size_t)i * HID * HID * 3, pD1[i], 16, 8, 3, HID * 3, 3, make_int4(0, 1, 2, 0));
        pack(dec_res_w2 + (size_t)i * HID * HID,     pD2[i], 16, 8, 1, HID,     1, make_int4(0, 0, 0, 0));
    }
    pack(enc_final_w, pEF, 4, 8, 3, HID * 3, 3, make_int4(0, 1, 2, 0));
    pack(dec_convT_w, pDTe, 16, 2, 2, 4, HID * 4, make_int4(1, 3, 0, 0));   // even outputs
    pack(dec_convT_w, pDTo, 16, 2, 2, 4, HID * 4, make_int4(0, 2, 0, 0));   // odd outputs
    pack(out_proj_w, pW, 512, 256, 1, NITEMS, 1, make_int4(0, 0, 0, 0));
    for (int q = 0; q < NQ; ++q)
        pack(codebooks + (size_t)q * NEMB * LAT, pCB + (size_t)q * 32768,
             32, 2, 1, LAT, 1, make_int4(0, 0, 0, 0));
    cb_norms_kernel<<<(NQ * NEMB) / 256, 256, 0, stream>>>(codebooks, cbNorms);

    // encoder stem: gather + conv(1->256,k4,s2,p1)+relu -> bufA [B][4096][256]
    enc_conv_kernel<<<(B_ * LENC * 32) / 256, 256, 0, stream>>>(
        user_ids, imat, enc_conv_w, enc_conv_b, bufA);

    dim3 blk(256);
    auto smem0 = [](int NT, int G) {   // staging vs bounce, OUTMODE 0
        size_t a = (size_t)2 * NT * G * 1024, c = (size_t)4096 * G;
        return (a > c) ? a : c;
    };
    // encoder residual blocks (L=4096)
    for (int i = 0; i < 2; ++i) {
        conv_wmma_kernel<16, 0><<<dim3(LENC / 128, 1, B_), blk, smem0(3, 16), stream>>>(
            bufA, pE1[i], enc_res_b1 + i * HID, nullptr, bufB,
            HID, LENC, LENC, HID, 3, make_int4(-1, 0, 1, 0), 1, 0, 1, 1);
        conv_wmma_kernel<16, 0><<<dim3(LENC / 128, 1, B_), blk, smem0(1, 16), stream>>>(
            bufB, pE2[i], enc_res_b2 + i * HID, bufA, bufA,
            HID, LENC, LENC, HID, 1, make_int4(0, 0, 0, 0), 1, 0, 1, 1);
    }
    // enc_final (256->64, k3) -> residual f32 in [B][LAT][L] (VQ row layout)
    conv_wmma_kernel<4, 1><<<dim3(LENC / 128, 1, B_), blk, 2 * 3 * 4 * 1024, stream>>>(
        bufA, pEF, enc_final_b, nullptr, residual,
        HID, LENC, LENC, LAT, 3, make_int4(-1, 0, 1, 0), 1, 0, 0, 1);

    // residual VQ, 4 stages (WMMA distance search)
    for (int q = 0; q < NQ; ++q) {
        vq_wmma_kernel<<<MROWS / 128, 256, 69632, stream>>>(
            codebooks + (size_t)q * NEMB * LAT, pCB + (size_t)q * 32768,
            cbNorms + q * NEMB, residual, zq, counts + q * NEMB, lossAcc);
    }
    zq_to_nlc_kernel<<<(B_ * LAT * LENC) / 256, 256, 0, stream>>>(zq, zqN);

    // decoder convT (64->256, k4, s2, p1) split into output parities
    conv_wmma_kernel<16, 0><<<dim3(LENC / 128, 1, B_), blk, smem0(2, 16), stream>>>(
        zqN, pDTe, dec_convT_b, nullptr, bufA,
        LAT, LENC, NITEMS, HID, 2, make_int4(0, -1, 0, 0), 2, 0, 1, 1);
    conv_wmma_kernel<16, 0><<<dim3(LENC / 128, 1, B_), blk, smem0(2, 16), stream>>>(
        zqN, pDTo, dec_convT_b, nullptr, bufA,
        LAT, LENC, NITEMS, HID, 2, make_int4(1, 0, 0, 0), 2, 1, 1, 1);

    // decoder residual blocks (L=8192)
    for (int i = 0; i < 2; ++i) {
        conv_wmma_kernel<16, 0><<<dim3(NITEMS / 128, 1, B_), blk, smem0(3, 16), stream>>>(
            bufA, pD1[i], dec_res_b1 + i * HID, nullptr, bufB,
            HID, NITEMS, NITEMS, HID, 3, make_int4(-1, 0, 1, 0), 1, 0, 1, 1);
        conv_wmma_kernel<16, 0><<<dim3(NITEMS / 128, 1, B_), blk, smem0(1, 16), stream>>>(
            bufB, pD2[i], dec_res_b2 + i * HID, bufA, bufA,
            HID, NITEMS, NITEMS, HID, 1, make_int4(0, 0, 0, 0), 1, 0, 1, 1);
    }

    // dec finalT (256->1) -> bf16 rows for the projection GEMM
    dec_finalT_kernel<<<(B_ * NITEMS) / 256, 256, 0, stream>>>(
        bufA, dec_fin_w, dec_fin_b, dbf);

    // recon = d @ W^T + b : split-K over 8 partials, then reduce + bias
    conv_wmma_kernel<16, 2><<<dim3(1, NITEMS / 256, KSPLIT), blk, 2 * 1 * 16 * 1024, stream>>>(
        dbf, pW, nullptr, nullptr, partOut,
        NITEMS, B_, B_, NITEMS, 1, make_int4(0, 0, 0, 0), 1, 0, 0, KSPLIT);
    reduce_out_kernel<<<(B_ * NITEMS) / 256, 256, 0, stream>>>(partOut, out_proj_b, out);

    finalize_kernel<<<1, 256, 0, stream>>>(counts, lossAcc, out + (size_t)B_ * NITEMS);
}
